// Llama3Attention_47828755808332
// MI455X (gfx1250) — matrix operand order repaired and checked by reading, not compiled, not run on hardware
//
#include <hip/hip_runtime.h>

// ---------------- problem constants ----------------
#define SEQ    2048
#define HIDDEN 4096
#define NH     32
#define NKV    8
#define HD     128
// softmax scale 1/sqrt(128)
#define SM_SCALE 0.08838834764831845f
// log2(500000)
#define LOG2_THETA 18.931568569324174f

typedef __attribute__((ext_vector_type(16))) __bf16 v16bf;
typedef __attribute__((ext_vector_type(8)))  float  v8f;

// fp32 -> bf16 (round to nearest even), stored as u16
__device__ __forceinline__ unsigned short f2bf(float f) {
    unsigned u = __float_as_uint(f);
    u += 0x7FFFu + ((u >> 16) & 1u);
    return (unsigned short)(u >> 16);
}

union FragU { uint4 q[2]; v16bf v; };

// build a 16-element bf16 fragment from two 16-byte chunks
__device__ __forceinline__ v16bf ld_frag2(const unsigned short* p0, const unsigned short* p1) {
    FragU t;
    t.q[0] = *(const uint4*)p0;
    t.q[1] = *(const uint4*)p1;
    return t.v;
}
__device__ __forceinline__ v16bf ld_frag1(const unsigned short* p) {
    return ld_frag2(p, p + 8);
}

// ---- CDNA5 async memory->LDS copy (ASYNCcnt-tracked, bypasses VGPRs) ----
// VDST holds the LDS byte address (low 32 bits of the generic pointer per
// the LDS aperture mapping); VADDR is the 64-bit
// global address.
__device__ __forceinline__ void async_b128(const void* gptr, void* lptr) {
    unsigned lds = (unsigned)(unsigned long long)lptr;
    asm volatile("global_load_async_to_lds_b128 %0, %1, off"
                 :: "v"(lds), "v"(gptr)
                 : "memory");
}
__device__ __forceinline__ void wait_async0() {
#if __has_builtin(__builtin_amdgcn_s_wait_asynccnt)
    __builtin_amdgcn_s_wait_asynccnt(0);
#else
    asm volatile("s_wait_asynccnt 0x0" ::: "memory");
#endif
}

// ---------------- fp32 -> bf16 convert ----------------
__global__ void cvt_f32_bf16(const float* __restrict__ src,
                             unsigned short* __restrict__ dst, int n) {
    int i = blockIdx.x * blockDim.x + threadIdx.x;
    if (i < n) dst[i] = f2bf(src[i]);
}

// ---------------- bf16 GEMM: C[M][N] = A[M][K] * W[N][K]^T ----------------
// block tile 128x128, 256 threads = 8 waves (wave32), K step 32.
// Double-buffered LDS staging via global_load_async_to_lds_b128 so the copy
// for step i+1 overlaps the 8 WMMAs of step i.  All fragments for a K-step
// are preloaded into distinct registers so the WMMA chain runs without
// per-pair s_wait_dscnt stalls.
__global__ __launch_bounds__(256) void gemm_bf16_nt(
    const unsigned short* __restrict__ A,
    const unsigned short* __restrict__ W,
    float* __restrict__ C,
    int M, int N, int K)
{
    __shared__ __align__(16) unsigned short ldsA[2][128 * 32];
    __shared__ __align__(16) unsigned short ldsB[2][128 * 32];

    const int tid  = threadIdx.x;
    const int wave = tid >> 5;
    const int lane = tid & 31;
    const int r    = lane & 15;
    const int half = lane >> 4;

    const int m0 = blockIdx.y * 128;
    const int n0 = blockIdx.x * 128;
    const int wm = (wave & 3) * 32;
    const int wn = (wave >> 2) * 64;

    v8f acc[2][4];
#pragma unroll
    for (int i = 0; i < 2; i++)
#pragma unroll
        for (int j = 0; j < 4; j++) acc[i][j] = (v8f){};

    const int lrow = tid >> 1;          // 0..127
    const int lseg = (tid & 1) * 16;    // 0 or 16 (ushort units)

    const unsigned short* aRow = &A[(size_t)(m0 + lrow) * K + lseg];
    const unsigned short* wRow = &W[(size_t)(n0 + lrow) * K + lseg];
    unsigned short* la = &ldsA[0][lrow * 32 + lseg];
    unsigned short* lb = &ldsB[0][lrow * 32 + lseg];

    // stage K-tile k0 into buffer buf (4 async b128 per thread)
    auto stage = [&](int k0, int buf) {
        async_b128(aRow + k0,     la + buf * (128 * 32));
        async_b128(aRow + k0 + 8, la + buf * (128 * 32) + 8);
        async_b128(wRow + k0,     lb + buf * (128 * 32));
        async_b128(wRow + k0 + 8, lb + buf * (128 * 32) + 8);
    };

    const int nsteps = K / 32;
    stage(0, 0);

    for (int step = 0; step < nsteps; step++) {
        const int buf = step & 1;
        wait_async0();          // this wave's copies for `buf` are done
        __syncthreads();        // => every wave's copies for `buf` are done
        if (step + 1 < nsteps)
            stage((step + 1) * 32, buf ^ 1);   // overlaps compute below

        const unsigned short* bufA = ldsA[buf];
        const unsigned short* bufB = ldsB[buf];

        // Preload ALL fragments for this K-step into distinct registers:
        // A: lane holds row (lane%16), K=16 per half
        // B: lane holds column n = lane%16 (a row of W), K=16 per half
        v16bf af[2], bfr[4];
#pragma unroll
        for (int mt = 0; mt < 2; mt++) {
            const unsigned short* base = &bufA[(wm + mt * 16 + r) * 32];
            af[mt] = ld_frag1(base + half * 16);
        }
#pragma unroll
        for (int nt = 0; nt < 4; nt++)
            bfr[nt] = ld_frag1(&bufB[(wn + nt * 16 + r) * 32 + half * 16]);

#pragma unroll
        for (int nt = 0; nt < 4; nt++)
#pragma unroll
            for (int mt = 0; mt < 2; mt++)
                acc[mt][nt] = __builtin_amdgcn_wmma_f32_16x16x32_bf16(
                    false, af[mt], false, bfr[nt], (short)0, acc[mt][nt], false, false);

        __syncthreads();        // protect buf from next iteration's overwrite
    }

    // C/D layout: VGPR j -> row j (lanes 0-15) / row 8+j (lanes 16-31);
    // col = lane%16.  One base address per (mt,j); nt-stores use immediate
    // offsets to avoid per-element 64-bit address chains.
#pragma unroll
    for (int mt = 0; mt < 2; mt++)
#pragma unroll
        for (int j = 0; j < 8; j++) {
            int row = m0 + wm + mt * 16 + j + half * 8;
            float* p = &C[(size_t)row * N + n0 + wn + r];
#pragma unroll
            for (int nt = 0; nt < 4; nt++)
                p[nt * 16] = acc[mt][nt][j];
        }
}

// ---------------- RoPE + repack to head-major bf16 ----------------
// src: fp32 [SEQ][H*HD]  ->  dst: bf16 [H][SEQ][HD]
__global__ void rope_pack(const float* __restrict__ src,
                          unsigned short* __restrict__ dst,
                          int H, int applyRope, float outScale)
{
    int i = blockIdx.x * blockDim.x + threadIdx.x;
    int total = SEQ * H * (HD / 2);
    if (i >= total) return;
    int d = i % (HD / 2);
    int h = (i / (HD / 2)) % H;
    int s = i / ((HD / 2) * H);

    const float* p = src + (size_t)s * (H * HD) + h * HD;
    float x1 = p[d], x2 = p[d + HD / 2];
    float o1, o2;
    if (applyRope) {
        float inv_freq = exp2f(-(float)d * (LOG2_THETA / 64.0f));
        float ang = (float)s * inv_freq;
        float sn, cs;
        __sincosf(ang, &sn, &cs);
        o1 = x1 * cs - x2 * sn;
        o2 = x2 * cs + x1 * sn;
    } else {
        o1 = x1; o2 = x2;
    }
    unsigned short* q = dst + ((size_t)h * SEQ + s) * HD;
    q[d]          = f2bf(o1 * outScale);
    q[d + HD / 2] = f2bf(o2 * outScale);
}

// ---------------- flash attention (causal, GQA) ----------------
// 1 workgroup = (head, 128 query rows); 8 waves x 16 rows. Key blocks of 32.
__global__ __launch_bounds__(256) void flash_attn(
    const unsigned short* __restrict__ Qb,  // [NH][SEQ][HD], pre-scaled by 1/sqrt(d)
    const unsigned short* __restrict__ Kb,  // [NKV][SEQ][HD]
    const unsigned short* __restrict__ Vb,  // [NKV][SEQ][HD]
    unsigned short* __restrict__ Ob)        // [SEQ][NH*HD]
{
    __shared__ __align__(16) unsigned short ldsK[32 * HD];   // [key][d]
    __shared__ __align__(16) unsigned short ldsV[HD * 32];   // [d][key] (transposed)
    __shared__ __align__(16) unsigned short ldsP[8][16][32]; // per-wave P tile

    const int h    = blockIdx.y;
    const int kvh  = h >> 2;                 // 4 Q heads per KV head
    const int qb0  = blockIdx.x * 128;
    const int tid  = threadIdx.x;
    const int wave = tid >> 5;
    const int lane = tid & 31;
    const int r    = lane & 15;
    const int half = lane >> 4;
    const int qrow = qb0 + wave * 16;

    // Q A-fragments for d = 0..127 (4 chunks of K=32), held in registers
    v16bf qf[4];
    {
        const unsigned short* qp = Qb + ((size_t)h * SEQ + qrow + r) * HD;
#pragma unroll
        for (int kk = 0; kk < 4; kk++)
            qf[kk] = ld_frag1(qp + kk * 32 + half * 16);
    }

    v8f o[8];
#pragma unroll
    for (int i = 0; i < 8; i++) o[i] = (v8f){};
    float mrow[8], lrow[8];
#pragma unroll
    for (int j = 0; j < 8; j++) { mrow[j] = -3.0e38f; lrow[j] = 0.0f; }

    const int nkb  = qb0 / 32 + 4;           // causal: keys up to qb0+127
    const int lkey = tid >> 3;               // 0..31
    const int lseg = (tid & 7) * 16;         // 0..112

    for (int kb = 0; kb < nkb; kb++) {
        const int k0 = kb * 32;
        __syncthreads();
        // stage K via async copy (no VGPR round trip), V transposed manually
        {
            const unsigned short* kp = Kb + ((size_t)kvh * SEQ + k0 + lkey) * HD + lseg;
            async_b128(kp,     &ldsK[lkey * HD + lseg]);
            async_b128(kp + 8, &ldsK[lkey * HD + lseg + 8]);

            const unsigned short* vp = Vb + ((size_t)kvh * SEQ + k0 + lkey) * HD + lseg;
            uint4 va = *(const uint4*)vp;
            uint4 vb = *(const uint4*)(vp + 8);
            const unsigned short* ea = (const unsigned short*)&va;
            const unsigned short* eb = (const unsigned short*)&vb;
#pragma unroll
            for (int e = 0; e < 8; e++) ldsV[(lseg + e) * 32 + lkey]     = ea[e];
#pragma unroll
            for (int e = 0; e < 8; e++) ldsV[(lseg + 8 + e) * 32 + lkey] = eb[e];
        }
        wait_async0();
        __syncthreads();

        // S = Q * K^T : two 16-key tiles; preload the 4 K-chunk fragments so
        // the 4-deep WMMA accumulation chain issues without ds waits between.
        v8f sacc[2];
#pragma unroll
        for (int nt = 0; nt < 2; nt++) {
            v16bf kf4[4];
#pragma unroll
            for (int kk = 0; kk < 4; kk++)
                kf4[kk] = ld_frag1(&ldsK[(nt * 16 + r) * HD + kk * 32 + half * 16]);
            v8f a = (v8f){};
#pragma unroll
            for (int kk = 0; kk < 4; kk++)
                a = __builtin_amdgcn_wmma_f32_16x16x32_bf16(
                        false, qf[kk], false, kf4[kk], (short)0, a, false, false);
            sacc[nt] = a;
        }

        // causal mask (scale already folded into Q)
#pragma unroll
        for (int nt = 0; nt < 2; nt++) {
            int kcol = k0 + nt * 16 + r;
#pragma unroll
            for (int j = 0; j < 8; j++) {
                int q = qrow + j + half * 8;
                if (kcol > q) sacc[nt][j] = -3.0e38f;
            }
        }

        // online softmax; row j lives in lanes of one 16-lane half
        float alpha[8];
#pragma unroll
        for (int j = 0; j < 8; j++) {
            float t = fmaxf(sacc[0][j], sacc[1][j]);
#pragma unroll
            for (int msk = 8; msk >= 1; msk >>= 1) t = fmaxf(t, __shfl_xor(t, msk, 32));
            float mn = fmaxf(mrow[j], t);
            alpha[j] = __expf(mrow[j] - mn);
            float p0 = __expf(sacc[0][j] - mn);
            float p1 = __expf(sacc[1][j] - mn);
            sacc[0][j] = p0; sacc[1][j] = p1;
            float rs = p0 + p1;
#pragma unroll
            for (int msk = 8; msk >= 1; msk >>= 1) rs += __shfl_xor(rs, msk, 32);
            lrow[j] = lrow[j] * alpha[j] + rs;
            mrow[j] = mn;
        }
#pragma unroll
        for (int dt = 0; dt < 8; dt++)
#pragma unroll
            for (int j = 0; j < 8; j++) o[dt][j] *= alpha[j];

        // P: C-layout -> A-layout via per-wave LDS round trip
#pragma unroll
        for (int j = 0; j < 8; j++) {
            ldsP[wave][j + half * 8][r]      = f2bf(sacc[0][j]);
            ldsP[wave][j + half * 8][r + 16] = f2bf(sacc[1][j]);
        }
        const unsigned short* pb = &ldsP[wave][r][0];
        v16bf pf = ld_frag1(pb + half * 16);

        // O += P * V (8 d-tiles of 16; independent accumulators let the
        // scheduler overlap ds loads with the XDL pipe)
#pragma unroll
        for (int dt = 0; dt < 8; dt++) {
            const unsigned short* vl = &ldsV[(dt * 16 + r) * 32 + half * 16];
            v16bf vf = ld_frag1(vl);
            o[dt] = __builtin_amdgcn_wmma_f32_16x16x32_bf16(
                        false, pf, false, vf, (short)0, o[dt], false, false);
        }
    }

    // normalize and store bf16 O in [SEQ][NH*HD] for the output GEMM
#pragma unroll
    for (int j = 0; j < 8; j++) {
        float inv = 1.0f / lrow[j];
        int srow = qrow + j + half * 8;
        unsigned short* op = Ob + (size_t)srow * (NH * HD) + h * HD + r;
#pragma unroll
        for (int dt = 0; dt < 8; dt++)
            op[dt * 16] = f2bf(o[dt][j] * inv);
    }
}

// ---------------- host-side orchestration ----------------
extern "C" void kernel_launch(void* const* d_in, const int* in_sizes, int n_in,
                              void* d_out, int out_size, void* d_ws, size_t ws_size,
                              hipStream_t stream) {
    (void)in_sizes; (void)n_in; (void)out_size; (void)ws_size;
    const float* X  = (const float*)d_in[0]; // [SEQ][HIDDEN]
    const float* Wq = (const float*)d_in[1]; // [4096][4096]
    const float* Wk = (const float*)d_in[2]; // [1024][4096]
    const float* Wv = (const float*)d_in[3]; // [1024][4096]
    const float* Wo = (const float*)d_in[4]; // [4096][4096]
    float* Y = (float*)d_out;                // [SEQ][HIDDEN]

    char* ws = (char*)d_ws;
    size_t off = 0;
    auto take = [&](size_t bytes) -> void* {
        void* p = (void*)(ws + off);
        off += (bytes + 255) & ~(size_t)255;
        return p;
    };

    unsigned short* Xb  = (unsigned short*)take((size_t)SEQ * HIDDEN * 2);
    unsigned short* Wqb = (unsigned short*)take((size_t)HIDDEN * HIDDEN * 2);
    unsigned short* Wkb = (unsigned short*)take((size_t)(NKV * HD) * HIDDEN * 2);
    unsigned short* Wvb = (unsigned short*)take((size_t)(NKV * HD) * HIDDEN * 2);
    unsigned short* Wob = (unsigned short*)take((size_t)HIDDEN * HIDDEN * 2);
    float* Qf = (float*)take((size_t)SEQ * HIDDEN * 4);
    float* Kf = (float*)take((size_t)SEQ * (NKV * HD) * 4);
    float* Vf = (float*)take((size_t)SEQ * (NKV * HD) * 4);
    unsigned short* Qb = (unsigned short*)take((size_t)NH  * SEQ * HD * 2);
    unsigned short* Kb = (unsigned short*)take((size_t)NKV * SEQ * HD * 2);
    unsigned short* Vb = (unsigned short*)take((size_t)NKV * SEQ * HD * 2);
    unsigned short* Ob = (unsigned short*)take((size_t)SEQ * HIDDEN * 2);

    // 1) fp32 -> bf16
    {
        int n;
        n = SEQ * HIDDEN;
        cvt_f32_bf16<<<(n + 255) / 256, 256, 0, stream>>>(X, Xb, n);
        n = HIDDEN * HIDDEN;
        cvt_f32_bf16<<<(n + 255) / 256, 256, 0, stream>>>(Wq, Wqb, n);
        cvt_f32_bf16<<<(n + 255) / 256, 256, 0, stream>>>(Wo, Wob, n);
        n = (NKV * HD) * HIDDEN;
        cvt_f32_bf16<<<(n + 255) / 256, 256, 0, stream>>>(Wk, Wkb, n);
        cvt_f32_bf16<<<(n + 255) / 256, 256, 0, stream>>>(Wv, Wvb, n);
    }

    // 2) QKV projections (bf16 WMMA, f32 out)
    gemm_bf16_nt<<<dim3(HIDDEN / 128, SEQ / 128), 256, 0, stream>>>(
        Xb, Wqb, Qf, SEQ, HIDDEN, HIDDEN);
    gemm_bf16_nt<<<dim3((NKV * HD) / 128, SEQ / 128), 256, 0, stream>>>(
        Xb, Wkb, Kf, SEQ, NKV * HD, HIDDEN);
    gemm_bf16_nt<<<dim3((NKV * HD) / 128, SEQ / 128), 256, 0, stream>>>(
        Xb, Wvb, Vf, SEQ, NKV * HD, HIDDEN);

    // 3) RoPE + repack head-major bf16 (fold softmax scale into Q)
    {
        int n;
        n = SEQ * NH * (HD / 2);
        rope_pack<<<(n + 255) / 256, 256, 0, stream>>>(Qf, Qb, NH, 1, SM_SCALE);
        n = SEQ * NKV * (HD / 2);
        rope_pack<<<(n + 255) / 256, 256, 0, stream>>>(Kf, Kb, NKV, 1, 1.0f);
        rope_pack<<<(n + 255) / 256, 256, 0, stream>>>(Vf, Vb, NKV, 0, 1.0f);
    }

    // 4) causal GQA flash attention
    flash_attn<<<dim3(SEQ / 128, NH), 256, 0, stream>>>(Qb, Kb, Vb, Ob);

    // 5) output projection -> fp32 d_out
    gemm_bf16_nt<<<dim3(HIDDEN / 128, SEQ / 128), 256, 0, stream>>>(
        Ob, Wob, Y, SEQ, HIDDEN, HIDDEN);
}
